// Extractor_62818191671633
// MI455X (gfx1250) — matrix operand order repaired and checked by reading, not compiled, not run on hardware
//
#include <hip/hip_runtime.h>

typedef __attribute__((ext_vector_type(16))) __bf16 v16bf;
typedef __attribute__((ext_vector_type(8)))  __bf16 v8bf;
typedef __attribute__((ext_vector_type(4)))  __bf16 v4bf;
typedef __attribute__((ext_vector_type(8)))  float  v8f;
typedef __attribute__((ext_vector_type(4)))  float  f32x4;
typedef __attribute__((ext_vector_type(4)))  int    v4i;

#define LDA 40   // __bf16 per A row in LDS (32 + pad); row = 80B, 16B-aligned
#define LDB 40   // __bf16 per B row in LDS (32 + pad); row = 80B, 16B-aligned

#if __has_builtin(__builtin_amdgcn_global_load_async_to_lds_b128) && \
    __has_builtin(__builtin_amdgcn_s_wait_asynccnt)
#define HAVE_ASYNC 1
#else
#define HAVE_ASYNC 0
#endif

// ---------------------------------------------------------------------------
// Global average pool: one block per (b, c) row; out = mean over H*W.
// ---------------------------------------------------------------------------
__global__ void pool_kernel(const float* __restrict__ x, float* __restrict__ pooled, int HW) {
    int row = blockIdx.x;
    const f32x4* p = (const f32x4*)(x + (size_t)row * HW);
    float s = 0.f;
    for (int i = threadIdx.x; i < HW / 4; i += blockDim.x) {
        f32x4 v = p[i];
        s += v[0] + v[1] + v[2] + v[3];
    }
    __shared__ float red[256];
    red[threadIdx.x] = s;
    __syncthreads();
    for (int off = 128; off > 0; off >>= 1) {
        if ((int)threadIdx.x < off) red[threadIdx.x] += red[threadIdx.x + off];
        __syncthreads();
    }
    if (threadIdx.x == 0) pooled[row] = red[0] / (float)HW;
}

// ---------------------------------------------------------------------------
// Attention MLP + softmax (K=4) + aggregated per-sample bias. One block per b.
// ---------------------------------------------------------------------------
__global__ void attention_kernel(const float* __restrict__ pooled,  // [B,C]
                                 const float* __restrict__ w1,      // [Hd,C]
                                 const float* __restrict__ w2,      // [4,Hd]
                                 const float* __restrict__ b2,      // [4]
                                 const float* __restrict__ bias,    // [4,Co]
                                 float* __restrict__ att,           // [B,4]
                                 float* __restrict__ aggb,          // [B,Co]
                                 int C, int Hd, int Co) {
    int b = blockIdx.x;
    int t = threadIdx.x;
    __shared__ float h[64];
    __shared__ float a[4];
    if (t < Hd) {
        float s = 0.f;
        for (int c = 0; c < C; ++c) s += pooled[b * C + c] * w1[t * C + c];
        h[t] = fmaxf(s, 0.f);
    }
    __syncthreads();
    if (t == 0) {
        float z[4]; float zmax = -1e30f;
        for (int k = 0; k < 4; ++k) {
            float s = b2[k];
            for (int j = 0; j < Hd; ++j) s += h[j] * w2[k * Hd + j];
            z[k] = s; zmax = fmaxf(zmax, s);
        }
        float den = 0.f;
        for (int k = 0; k < 4; ++k) { z[k] = __expf(z[k] - zmax); den += z[k]; }
        for (int k = 0; k < 4; ++k) { a[k] = z[k] / den; att[b * 4 + k] = a[k]; }
    }
    __syncthreads();
    for (int co = t; co < Co; co += blockDim.x) {
        float s = 0.f;
        for (int k = 0; k < 4; ++k) s += a[k] * bias[k * Co + co];
        aggb[b * Co + co] = s;
    }
}

// ---------------------------------------------------------------------------
// Aggregate K=4 expert kernels with per-sample attention -> bf16 weights.
// ---------------------------------------------------------------------------
__global__ void wagg_kernel(const float* __restrict__ att, const float* __restrict__ W,
                            __bf16* __restrict__ wagg, int perB, int total) {
    int i = blockIdx.x * blockDim.x + threadIdx.x;
    if (i >= total) return;
    int b = i / perB;
    int r = i - b * perB;
    const float* a = att + b * 4;
    float v = a[0] * W[r] + a[1] * W[(size_t)perB + r] +
              a[2] * W[2 * (size_t)perB + r] + a[3] * W[3 * (size_t)perB + r];
    wagg[i] = (__bf16)v;
}

// ---------------------------------------------------------------------------
// Implicit-GEMM dynamic conv, per-sample bf16 weights, WMMA bf16, double-
// buffered LDS software pipeline. Block: 256 thr (8 waves).
// Tile: M=128 (all Co) x N=64 px (half output row). K-step = 32.
// ---------------------------------------------------------------------------
template <int KSZ>
__global__ __launch_bounds__(256) void dynconv_wmma_kernel(
    const float* __restrict__ in,     // [B, Cin, H, W] f32
    const __bf16* __restrict__ wagg,  // [B, Co, Cin*KSZ*KSZ] bf16
    const float* __restrict__ aggb,   // [B, Co]
    float* __restrict__ out,          // [B, Co, H, W]
    int Cin) {
    constexpr int PAD = KSZ / 2;
    constexpr int Co = 128, H = 128, W = 128;
    const int Ktot = Cin * KSZ * KSZ;

    const int tiles_per_row = W / 64;  // 2
    int bid = blockIdx.x;
    int b   = bid / (H * tiles_per_row);
    int rem = bid % (H * tiles_per_row);
    int h   = rem / tiles_per_row;
    int w0  = (rem % tiles_per_row) * 64;

    __shared__ alignas(16) __bf16 lds_a[2][128 * LDA];
    __shared__ alignas(16) __bf16 lds_b[2][64 * LDB];

    int tid  = threadIdx.x;
    int lane = tid & 31;
    int wave = tid >> 5;
    int m0 = (wave & 3) * 2;   // mtiles {m0, m0+1} (16 co each)
    int n0 = (wave >> 2) * 2;  // ntiles {n0, n0+1} (16 px each)

    const v8f vzero = {0.f, 0.f, 0.f, 0.f, 0.f, 0.f, 0.f, 0.f};
    v8f acc[2][2];
#pragma unroll
    for (int i = 0; i < 2; ++i)
#pragma unroll
        for (int j = 0; j < 2; ++j) acc[i][j] = vzero;

    const __bf16* wb = wagg + (size_t)b * Co * Ktot;
    const float*  xb = in + (size_t)b * Cin * H * W;

    // ---- staging state (register-resident between load & store phases) ----
    uint4  areg[2];
    __bf16 bq[2][4];

    auto loadA = [&](int k0) {
#pragma unroll
        for (int i = 0; i < 2; ++i) {
            int idx = tid + i * 256;  // 0..511 slots of 16B
            int co = idx >> 2, sl = idx & 3;
            areg[i] = *(const uint4*)((const char*)wb + ((size_t)co * Ktot + k0) * 2 + sl * 16);
        }
    };
    auto storeA = [&](int bufn) {
#pragma unroll
        for (int i = 0; i < 2; ++i) {
            int idx = tid + i * 256;
            int co = idx >> 2, sl = idx & 3;
            *(uint4*)&lds_a[bufn][co * LDA + sl * 8] = areg[i];
        }
    };
#if HAVE_ASYNC
    auto asyncA = [&](int k0, int bufn) {
#pragma unroll
        for (int i = 0; i < 2; ++i) {
            int idx = tid + i * 256;
            int co = idx >> 2, sl = idx & 3;
            const char* gp = (const char*)wb + ((size_t)co * Ktot + k0) * 2 + sl * 16;
            __bf16* lp = &lds_a[bufn][co * LDA + sl * 8];
            __attribute__((address_space(1))) v4i* gq =
                (__attribute__((address_space(1))) v4i*)const_cast<char*>(gp);
            __attribute__((address_space(3))) v4i* lq =
                (__attribute__((address_space(3))) v4i*)lp;
            __builtin_amdgcn_global_load_async_to_lds_b128(gq, lq, 0, 0);
        }
    };
#endif
    // B staging: each slot = one pixel column n x quad of 4 consecutive kk.
    // Global loads coalesce across lanes (consecutive n -> consecutive w);
    // LDS store is a single 8B v4bf per slot (k-contiguous, low bank conflict).
    auto loadB = [&](int k0) {
#pragma unroll
        for (int i = 0; i < 2; ++i) {
            int idx = tid + i * 256;   // 512 slots: 64 n x 8 k-quads
            int n = idx & 63, q = idx >> 6;
#pragma unroll
            for (int e = 0; e < 4; ++e) {
                int kg = k0 + q * 4 + e;
                int ci = kg / (KSZ * KSZ);
                int r2 = kg - ci * (KSZ * KSZ);
                int kh = r2 / KSZ, kw = r2 - kh * KSZ;
                int hi = h + kh - PAD;
                int wi = w0 + n + kw - PAD;
                float val = 0.f;
                if (hi >= 0 && hi < H && wi >= 0 && wi < W)
                    val = xb[((size_t)ci * H + hi) * W + wi];
                bq[i][e] = (__bf16)val;
            }
        }
    };
    auto storeB = [&](int bufn) {
#pragma unroll
        for (int i = 0; i < 2; ++i) {
            int idx = tid + i * 256;
            int n = idx & 63, q = idx >> 6;
            v4bf vv = {bq[i][0], bq[i][1], bq[i][2], bq[i][3]};
            *(v4bf*)&lds_b[bufn][n * LDB + q * 4] = vv;   // 8B-aligned
        }
    };

    // ---- prologue: stage K-step 0 into buffer 0 ----
#if HAVE_ASYNC
    asyncA(0, 0);
#else
    loadA(0);
#endif
    loadB(0);
#if !HAVE_ASYNC
    storeA(0);
#endif
    storeB(0);

    int buf = 0;
    for (int k0 = 0; k0 < Ktot; k0 += 32, buf ^= 1) {
#if HAVE_ASYNC
        __builtin_amdgcn_s_wait_asynccnt(0);
#endif
        __syncthreads();
        bool more = (k0 + 32) < Ktot;
        if (more) {
            // issue next-step global traffic early; it overlaps the WMMAs below
#if HAVE_ASYNC
            asyncA(k0 + 32, buf ^ 1);
#else
            loadA(k0 + 32);
#endif
            loadB(k0 + 32);
            if (k0 + 64 < Ktot)
                __builtin_prefetch((const char*)wb + (size_t)tid * Ktot * 2 + (size_t)(k0 + 64) * 2, 0, 1);
        }

        // ---- fragments from current buffer (CDNA5 WMMA lane layouts) ----
        int mrow = lane & 15;
        int half = lane >> 4;
        const __bf16* la = lds_a[buf];
        const __bf16* lb = lds_b[buf];
        v16bf afrag[2], bfrag[2];
#pragma unroll
        for (int i = 0; i < 2; ++i) {
            const v8bf* a0 = (const v8bf*)&la[((m0 + i) * 16 + mrow) * LDA + half * 8];
            const v8bf* a1 = (const v8bf*)&la[((m0 + i) * 16 + mrow) * LDA + 16 + half * 8];
            afrag[i] = __builtin_shufflevector(a0[0], a1[0], 0, 1, 2, 3, 4, 5, 6, 7,
                                               8, 9, 10, 11, 12, 13, 14, 15);
            const v8bf* bp = (const v8bf*)&lb[((n0 + i) * 16 + mrow) * LDB + half * 8];
            bfrag[i] = __builtin_shufflevector(bp[0], bp[2], 0, 1, 2, 3, 4, 5, 6, 7,
                                               8, 9, 10, 11, 12, 13, 14, 15);
        }
#pragma unroll
        for (int i = 0; i < 2; ++i)
#pragma unroll
            for (int j = 0; j < 2; ++j)
                acc[i][j] = __builtin_amdgcn_wmma_f32_16x16x32_bf16(
                    false, afrag[i], false, bfrag[j], (short)0, acc[i][j], false, false);

        if (more) {
#if !HAVE_ASYNC
            storeA(buf ^ 1);
#endif
            storeB(buf ^ 1);
        }
    }

    // ---- epilogue: D layout (VGPR r -> M = r + 8*half, N = lane%16) ----
    const float* bias = aggb + b * Co;
    float* ob = out + (((size_t)b * Co) * H + h) * W;
    int ncol = lane & 15;
    int hgrp = lane >> 4;
#pragma unroll
    for (int i = 0; i < 2; ++i)
#pragma unroll
        for (int j = 0; j < 2; ++j)
#pragma unroll
            for (int rr = 0; rr < 8; ++rr) {
                int co   = (m0 + i) * 16 + rr + 8 * hgrp;
                int wcol = w0 + (n0 + j) * 16 + ncol;
                ob[(size_t)co * H * W + wcol] = acc[i][j][rr] + bias[co];
            }
}

// ---------------------------------------------------------------------------
// BN stats pass 1: one block per (c, b) slice -> partial sum / sumsq.
// ---------------------------------------------------------------------------
__global__ void bn_partial_kernel(const float* __restrict__ y, float* __restrict__ partial,
                                  int C, int B, int HW) {
    int blk = blockIdx.x;          // c * B + b
    int c = blk / B, b = blk - c * B;
    const f32x4* p = (const f32x4*)(y + ((size_t)b * C + c) * HW);
    float s = 0.f, s2 = 0.f;
    for (int i = threadIdx.x; i < HW / 4; i += blockDim.x) {
        f32x4 v = p[i];
#pragma unroll
        for (int e = 0; e < 4; ++e) { s += v[e]; s2 += v[e] * v[e]; }
    }
    __shared__ float rs[256], rs2[256];
    rs[threadIdx.x] = s; rs2[threadIdx.x] = s2;
    __syncthreads();
    for (int off = 128; off > 0; off >>= 1) {
        if ((int)threadIdx.x < off) {
            rs[threadIdx.x]  += rs[threadIdx.x + off];
            rs2[threadIdx.x] += rs2[threadIdx.x + off];
        }
        __syncthreads();
    }
    if (threadIdx.x == 0) {
        partial[blk * 2]     = rs[0];
        partial[blk * 2 + 1] = rs2[0];
    }
}

// BN stats pass 2: fused scale/shift per channel.
__global__ void bn_finalize_kernel(const float* __restrict__ partial,
                                   const float* __restrict__ gamma,
                                   const float* __restrict__ beta,
                                   float* __restrict__ ab, int C, int B, int HW) {
    int c = blockIdx.x * blockDim.x + threadIdx.x;
    if (c >= C) return;
    float s = 0.f, s2 = 0.f;
    for (int b = 0; b < B; ++b) {
        s  += partial[(c * B + b) * 2];
        s2 += partial[(c * B + b) * 2 + 1];
    }
    float n = (float)B * (float)HW;
    float mean = s / n;
    float var  = s2 / n - mean * mean;
    float inv  = rsqrtf(var + 1e-5f);
    float a    = gamma[c] * inv;
    ab[c]      = a;
    ab[C + c]  = beta[c] - mean * a;
}

// ---------------------------------------------------------------------------
// In-place BN apply (+ optional leaky), vec4.
// ---------------------------------------------------------------------------
__global__ void bn_apply_kernel(float* __restrict__ y, const float* __restrict__ ab,
                                int C, int HW, size_t total4, int do_leaky) {
    size_t i = (size_t)blockIdx.x * blockDim.x + threadIdx.x;
    if (i >= total4) return;
    int c = (int)((i * 4 / HW) % C);
    float a = ab[c], sh = ab[C + c];
    f32x4 v = ((f32x4*)y)[i];
#pragma unroll
    for (int e = 0; e < 4; ++e) {
        float t = a * v[e] + sh;
        if (do_leaky) t = (t >= 0.f) ? t : 0.2f * t;
        v[e] = t;
    }
    ((f32x4*)y)[i] = v;
}

// ---------------------------------------------------------------------------
// Final: out = leaky( bn2(y2) + bns(r) ), vec4.
// ---------------------------------------------------------------------------
__global__ void final_kernel(const float* __restrict__ y2, const float* __restrict__ r,
                             const float* __restrict__ ab2, const float* __restrict__ abS,
                             float* __restrict__ out, int C, int HW, size_t total4) {
    size_t i = (size_t)blockIdx.x * blockDim.x + threadIdx.x;
    if (i >= total4) return;
    int c = (int)((i * 4 / HW) % C);
    float a2 = ab2[c], s2 = ab2[C + c], aS = abS[c], sS = abS[C + c];
    f32x4 vy = ((const f32x4*)y2)[i];
    f32x4 vr = ((const f32x4*)r)[i];
    f32x4 o;
#pragma unroll
    for (int e = 0; e < 4; ++e) {
        float t = a2 * vy[e] + s2 + aS * vr[e] + sS;
        o[e] = (t >= 0.f) ? t : 0.2f * t;
    }
    ((f32x4*)out)[i] = o;
}

// ---------------------------------------------------------------------------
extern "C" void kernel_launch(void* const* d_in, const int* in_sizes, int n_in,
                              void* d_out, int out_size, void* d_ws, size_t ws_size,
                              hipStream_t stream) {
    (void)in_sizes; (void)n_in; (void)out_size; (void)ws_size;
    const float* x    = (const float*)d_in[0];
    const float* a1w1 = (const float*)d_in[1];
    const float* a1w2 = (const float*)d_in[2];
    const float* a1b2 = (const float*)d_in[3];
    const float* w1   = (const float*)d_in[4];
    const float* b1   = (const float*)d_in[5];
    const float* g1   = (const float*)d_in[6];
    const float* be1  = (const float*)d_in[7];
    const float* a2w1 = (const float*)d_in[8];
    const float* a2w2 = (const float*)d_in[9];
    const float* a2b2 = (const float*)d_in[10];
    const float* w2   = (const float*)d_in[11];
    const float* b2   = (const float*)d_in[12];
    const float* g2   = (const float*)d_in[13];
    const float* be2  = (const float*)d_in[14];
    const float* asw1 = (const float*)d_in[15];
    const float* asw2 = (const float*)d_in[16];
    const float* asb2 = (const float*)d_in[17];
    const float* wsW  = (const float*)d_in[18];
    const float* bs   = (const float*)d_in[19];
    const float* gs   = (const float*)d_in[20];
    const float* bes  = (const float*)d_in[21];

    constexpr int B = 16, Ci = 64, Co = 128, H = 128, Wd = 128, HW = H * Wd;
    constexpr int H1 = 17, H2 = 33;
    constexpr size_t NBIG = (size_t)B * Co * HW;  // 33,554,432

    char* p = (char*)d_ws;
    auto alloc = [&](size_t bytes) -> char* {
        char* q = p;
        p += (bytes + 255) & ~(size_t)255;
        return q;
    };
    float*  y1      = (float*)alloc(NBIG * 4);
    float*  y2      = (float*)alloc(NBIG * 4);
    float*  rr      = (float*)alloc(NBIG * 4);
    __bf16* wagg1   = (__bf16*)alloc((size_t)B * Co * Ci * 9 * 2);
    __bf16* wagg2   = (__bf16*)alloc((size_t)B * Co * Co * 9 * 2);
    __bf16* waggS   = (__bf16*)alloc((size_t)B * Co * Ci * 2);
    float*  pooled1 = (float*)alloc(B * Ci * 4);
    float*  pooled2 = (float*)alloc(B * Co * 4);
    float*  att1    = (float*)alloc(B * 4 * 4);
    float*  att2    = (float*)alloc(B * 4 * 4);
    float*  attS    = (float*)alloc(B * 4 * 4);
    float*  aggb1   = (float*)alloc(B * Co * 4);
    float*  aggb2   = (float*)alloc(B * Co * 4);
    float*  aggbS   = (float*)alloc(B * Co * 4);
    float*  part1   = (float*)alloc(Co * B * 2 * 4);
    float*  part2   = (float*)alloc(Co * B * 2 * 4);
    float*  partS   = (float*)alloc(Co * B * 2 * 4);
    float*  bn1     = (float*)alloc(2 * Co * 4);
    float*  bn2     = (float*)alloc(2 * Co * 4);
    float*  bnS     = (float*)alloc(2 * Co * 4);

    const int conv_blocks = B * H * (Wd / 64);  // 4096
    const size_t ew4_blocks = (NBIG / 4 + 255) / 256;

    // attention paths for conv1 + shortcut (both use pooled(x))
    pool_kernel<<<B * Ci, 256, 0, stream>>>(x, pooled1, HW);
    attention_kernel<<<B, 128, 0, stream>>>(pooled1, a1w1, a1w2, a1b2, b1, att1, aggb1, Ci, H1, Co);
    attention_kernel<<<B, 128, 0, stream>>>(pooled1, asw1, asw2, asb2, bs, attS, aggbS, Ci, H1, Co);
    wagg_kernel<<<(B * Co * Ci * 9 + 255) / 256, 256, 0, stream>>>(att1, w1, wagg1, Co * Ci * 9, B * Co * Ci * 9);
    wagg_kernel<<<(B * Co * Ci + 255) / 256, 256, 0, stream>>>(attS, wsW, waggS, Co * Ci, B * Co * Ci);

    // conv1 -> bn1 -> leaky (in place)
    dynconv_wmma_kernel<3><<<conv_blocks, 256, 0, stream>>>(x, wagg1, aggb1, y1, Ci);
    bn_partial_kernel<<<Co * B, 256, 0, stream>>>(y1, part1, Co, B, HW);
    bn_finalize_kernel<<<1, Co, 0, stream>>>(part1, g1, be1, bn1, Co, B, HW);
    bn_apply_kernel<<<ew4_blocks, 256, 0, stream>>>(y1, bn1, Co, HW, NBIG / 4, 1);

    // attention path for conv2 (depends on y1n)
    pool_kernel<<<B * Co, 256, 0, stream>>>(y1, pooled2, HW);
    attention_kernel<<<B, 128, 0, stream>>>(pooled2, a2w1, a2w2, a2b2, b2, att2, aggb2, Co, H2, Co);
    wagg_kernel<<<(B * Co * Co * 9 + 255) / 256, 256, 0, stream>>>(att2, w2, wagg2, Co * Co * 9, B * Co * Co * 9);

    // conv2 and shortcut conv
    dynconv_wmma_kernel<3><<<conv_blocks, 256, 0, stream>>>(y1, wagg2, aggb2, y2, Co);
    dynconv_wmma_kernel<1><<<conv_blocks, 256, 0, stream>>>(x, waggS, aggbS, rr, Ci);

    // bn stats for both branches, fused final
    bn_partial_kernel<<<Co * B, 256, 0, stream>>>(y2, part2, Co, B, HW);
    bn_finalize_kernel<<<1, Co, 0, stream>>>(part2, g2, be2, bn2, Co, B, HW);
    bn_partial_kernel<<<Co * B, 256, 0, stream>>>(rr, partS, Co, B, HW);
    bn_finalize_kernel<<<1, Co, 0, stream>>>(partS, gs, bes, bnS, Co, B, HW);
    final_kernel<<<ew4_blocks, 256, 0, stream>>>(y2, rr, bn2, bnS, (float*)d_out, Co, HW, NBIG / 4);
}
